// ProGAT_73340861546804
// MI455X (gfx1250) — matrix operand order repaired and checked by reading, not compiled, not run on hardware
//
#include <hip/hip_runtime.h>

typedef __attribute__((ext_vector_type(16))) _Float16 v16h;
typedef __attribute__((ext_vector_type(8)))  _Float16 v8h;
typedef __attribute__((ext_vector_type(4)))  _Float16 v4h;
typedef __attribute__((ext_vector_type(8)))  float    v8f;
typedef __attribute__((ext_vector_type(4)))  float    v4f;
typedef __attribute__((ext_vector_type(2)))  float    v2f;

static constexpr int Bb    = 32;
static constexpr int Ss    = 512;
static constexpr int KNb   = 21;
static constexpr int Fin   = 26;
static constexpr int Ee    = 256;
static constexpr int Mrows = Bb * Ss;          // 16384
static constexpr int Mpad  = 128;              // padded row count for tiny GEMMs
static constexpr float NEGV = -900000000.0f;

static constexpr int TM = 128, TN = 128, TK = 32;

__device__ __forceinline__ float leakyf(float x) { return x > 0.f ? x : 0.01f * x; }

// fragment loads (one aligned 32B chunk per lane) + 8 WMMAs for a 32x64 wave tile
__device__ __forceinline__ void wmma_tiles(const _Float16* As, const _Float16* Bs,
                                           int wm, int wn, int lane, v8f acc[2][4])
{
  v16h afr[2], bfr[4];
#pragma unroll
  for (int mt = 0; mt < 2; ++mt)
    afr[mt] = *(const v16h*)&As[((wm * 2 + mt) * 32 + lane) * 16];
#pragma unroll
  for (int nt = 0; nt < 4; ++nt)
    bfr[nt] = *(const v16h*)&Bs[((wn * 4 + nt) * 32 + lane) * 16];
#pragma unroll
  for (int mt = 0; mt < 2; ++mt)
#pragma unroll
    for (int nt = 0; nt < 4; ++nt)
      acc[mt][nt] = __builtin_amdgcn_wmma_f32_16x16x32_f16(
          false, afr[mt], false, bfr[nt], (short)0, acc[mt][nt], false, false);
}

// ---------------------------------------------------------------------------
// gemm256h: C[M,N] = A16[M,256] @ W16[N,256]^T + bias[N], A/W already fp16.
// Tiles are DMA'd into LDS *in fragment order* with GLOBAL_LOAD_ASYNC_TO_LDS
// (each lane's 16-element chunk is 32 contiguous bytes in fp16 row-major
// global memory, so the swizzle is pure addressing). Double-buffered: tile
// ks+1's DMA is in flight during tile ks's WMMA block; s_wait_asynccnt 4
// retires the older 4 copies (async loads complete in order).
// M,N multiples of 128 -> no guards anywhere.
// ---------------------------------------------------------------------------
__global__ __launch_bounds__(256) void gemm256h_f16_wmma(
    const _Float16* __restrict__ A16, const _Float16* __restrict__ W16,
    const float* __restrict__ bias, float* __restrict__ C, int N)
{
  __shared__ alignas(32) _Float16 As[2][4096];   // 2 x 8KB
  __shared__ alignas(32) _Float16 Bs[2][4096];

  const int tid  = threadIdx.x;
  const int lane = tid & 31;
  const int wave = tid >> 5;
  const int wm   = wave >> 1;   // 0..3
  const int wn   = wave & 1;    // 0..1
  const int rowBase = blockIdx.x * TM;
  const int colBase = blockIdx.y * TN;
  const int half = lane >> 4, l15 = lane & 15;

  // per-lane source/dest for the async tile copies (this wave's tiles)
  // A frag: lane=(half,m&15); e0..15 <- k0+half*16..  (32B contiguous)
  // B frag: lane=(half,n&15); e0..15 <- k0+half*16..  (32B contiguous)
  const int rowA = rowBase + wave * 16 + l15;
  const int colB = colBase + wave * 16 + l15;
  const unsigned long long gA = (unsigned long long)(const void*)(A16 + (size_t)rowA * Ee + half * 16);
  const unsigned long long gB = (unsigned long long)(const void*)(W16 + (size_t)colB * Ee + half * 16);
  const unsigned ldsA = (unsigned)(unsigned long long)(const void*)&As[0][(wave * 32 + lane) * 16];
  const unsigned ldsB = (unsigned)(unsigned long long)(const void*)&Bs[0][(wave * 32 + lane) * 16];

  auto issue_tile = [&](int ks, int buf) {
    const unsigned long long gA0 = gA + (unsigned long long)ks * 64;  // 32 k * 2B
    const unsigned long long gA1 = gA0 + 16;
    const unsigned long long gB0 = gB + (unsigned long long)ks * 64;
    const unsigned la  = ldsA + (unsigned)buf * 8192;
    const unsigned la1 = la + 16;
    const unsigned lb  = ldsB + (unsigned)buf * 8192;
    asm volatile("global_load_async_to_lds_b128 %0, %1, off"
                 :: "v"(la),  "v"(gA0) : "memory");
    asm volatile("global_load_async_to_lds_b128 %0, %1, off"
                 :: "v"(la1), "v"(gA1) : "memory");
    asm volatile("global_load_async_to_lds_b128 %0, %1, off"
                 :: "v"(lb),  "v"(gB0) : "memory");
    asm volatile("global_load_async_to_lds_b128 %0, %1, off offset:16"
                 :: "v"(lb),  "v"(gB0) : "memory");   // IOFFSET adds to both sides
  };

  v8f acc[2][4];
  const v8f vzero = {0.f, 0.f, 0.f, 0.f, 0.f, 0.f, 0.f, 0.f};
#pragma unroll
  for (int i = 0; i < 2; ++i)
#pragma unroll
    for (int j = 0; j < 4; ++j) acc[i][j] = vzero;

  issue_tile(0, 0);
  for (int ks = 0; ks < 8; ++ks) {
    if (ks < 7) {
      issue_tile(ks + 1, (ks + 1) & 1);            // overlap DMA with WMMA
      asm volatile("s_wait_asynccnt 4" ::: "memory");  // tile ks's 4 copies done
    } else {
      asm volatile("s_wait_asynccnt 0" ::: "memory");
    }
    __syncthreads();
    wmma_tiles(&As[ks & 1][0], &Bs[ks & 1][0], wm, wn, lane, acc);
    __syncthreads();   // buffer (ks&1) free for tile ks+2's DMA
  }

  // ---- epilogue: bias + store (C/D layout per ISA 7.12.2), no guards ----
#pragma unroll
  for (int mt = 0; mt < 2; ++mt)
#pragma unroll
    for (int nt = 0; nt < 4; ++nt) {
      const int n  = colBase + wn * 64 + nt * 16 + (lane & 15);
      const float bv = bias[n];
#pragma unroll
      for (int rr = 0; rr < 8; ++rr) {
        const int m = rowBase + wm * 32 + mt * 16 + rr + 8 * (lane >> 4);
        C[(size_t)m * N + n] = acc[mt][nt][rr] + bv;
      }
    }
}

// ---------------------------------------------------------------------------
// gemm26: C[M,256] = leaky(A[M,26] @ W[256,26]^T + bias), plus fp16 mirror.
// Single K-step, k=26..31 zero-padded statically. Threads 0..127 stage one A
// row each, threads 128..255 one W column each. No runtime guards.
// ---------------------------------------------------------------------------
__global__ __launch_bounds__(256) void gemm26_f16_wmma(
    const float* __restrict__ A, const float* __restrict__ W,
    const float* __restrict__ bias, float* __restrict__ C,
    _Float16* __restrict__ C16)
{
  __shared__ alignas(32) _Float16 As[4096];
  __shared__ alignas(32) _Float16 Bs[4096];

  const int tid  = threadIdx.x;
  const int lane = tid & 31;
  const int wave = tid >> 5;
  const int wm   = wave >> 1;
  const int wn   = wave & 1;
  const int rowBase = blockIdx.x * TM;
  const int colBase = blockIdx.y * TN;

  {
    const int  r   = tid & 127;
    const bool isB = tid >= 128;
    const float* srcp = isB ? (W + (size_t)(colBase + r) * Fin)
                            : (A + (size_t)(rowBase + r) * Fin);
    float kv[32];
#pragma unroll
    for (int i = 0; i < 13; ++i) {        // rows are 104B -> 8B aligned
      v2f t = *(const v2f*)(srcp + 2 * i);
      kv[2 * i] = t[0]; kv[2 * i + 1] = t[1];
    }
#pragma unroll
    for (int i = 26; i < 32; ++i) kv[i] = 0.f;

    const int tt = r >> 4, lrow = r & 15;
    v8h h0a, h0b, h1a, h1b;
    if (!isB) {
#pragma unroll
      for (int e = 0; e < 8; ++e) {
        h0a[e] = (_Float16)kv[e];       h0b[e] = (_Float16)kv[8 + e];
        h1a[e] = (_Float16)kv[16 + e];  h1b[e] = (_Float16)kv[24 + e];
      }
      _Float16* d0 = &As[(tt * 32 + lrow) * 16];
      _Float16* d1 = &As[(tt * 32 + 16 + lrow) * 16];
      *(v8h*)d0 = h0a; *(v8h*)(d0 + 8) = h0b;
      *(v8h*)d1 = h1a; *(v8h*)(d1 + 8) = h1b;
    } else {
#pragma unroll
      for (int e = 0; e < 8; ++e) {
        h0a[e] = (_Float16)kv[e];       h0b[e] = (_Float16)kv[8 + e];
        h1a[e] = (_Float16)kv[16 + e];  h1b[e] = (_Float16)kv[24 + e];
      }
      _Float16* d0 = &Bs[(tt * 32 + lrow) * 16];
      _Float16* d1 = &Bs[(tt * 32 + 16 + lrow) * 16];
      *(v8h*)d0 = h0a; *(v8h*)(d0 + 8) = h0b;
      *(v8h*)d1 = h1a; *(v8h*)(d1 + 8) = h1b;
    }
  }
  __syncthreads();

  v8f acc[2][4];
  const v8f vzero = {0.f, 0.f, 0.f, 0.f, 0.f, 0.f, 0.f, 0.f};
#pragma unroll
  for (int i = 0; i < 2; ++i)
#pragma unroll
    for (int j = 0; j < 4; ++j) acc[i][j] = vzero;
  wmma_tiles(As, Bs, wm, wn, lane, acc);

#pragma unroll
  for (int mt = 0; mt < 2; ++mt)
#pragma unroll
    for (int nt = 0; nt < 4; ++nt) {
      const int n  = colBase + wn * 64 + nt * 16 + (lane & 15);
      const float bv = bias[n];
#pragma unroll
      for (int rr = 0; rr < 8; ++rr) {
        const int m = rowBase + wm * 32 + mt * 16 + rr + 8 * (lane >> 4);
        const float v = leakyf(acc[mt][nt][rr] + bv);
        C[(size_t)m * Ee + n]   = v;
        C16[(size_t)m * Ee + n] = (_Float16)v;
      }
    }
}

// ---------------------------------------------------------------------------
// fp32 -> fp16 bulk convert (4 elems/thread); n4 = count/4
// ---------------------------------------------------------------------------
__global__ __launch_bounds__(256) void cvt16_kernel(
    const float* __restrict__ src, _Float16* __restrict__ dst, int n4)
{
  const int i = blockIdx.x * 256 + threadIdx.x;
  if (i < n4) {
    const v4f v = *((const v4f*)src + i);
    v4h h;
#pragma unroll
    for (int j = 0; j < 4; ++j) h[j] = (_Float16)v[j];
    *((v4h*)dst + i) = h;
  }
}

// ---------------------------------------------------------------------------
// per-row dot with an E-vector: out[row] = dot(X[row,:], w); one wave32/row
// ---------------------------------------------------------------------------
__global__ __launch_bounds__(256) void rowdot_kernel(
    const float* __restrict__ X, const float* __restrict__ w,
    float* __restrict__ out, int Mr)
{
  const int row  = blockIdx.x * 8 + (threadIdx.x >> 5);
  const int lane = threadIdx.x & 31;
  if (row >= Mr) return;
  float s = 0.f;
#pragma unroll
  for (int c = 0; c < Ee; c += 32) s += X[(size_t)row * Ee + c + lane] * w[c + lane];
#pragma unroll
  for (int o = 16; o; o >>= 1) s += __shfl_xor(s, o, 32);
  if (lane == 0) out[row] = s;
}

// ---------------------------------------------------------------------------
// Neighbor attention: scores -> softmax over K=21 -> gather-weighted sum of
// t rows -> elu -> fp16 context. One wave32 per node.
// ---------------------------------------------------------------------------
__global__ __launch_bounds__(256) void attend_ctx_kernel(
    const int* __restrict__ idx, const float* __restrict__ aA,
    const float* __restrict__ gB, const float* __restrict__ alignb,
    const float* __restrict__ t, _Float16* __restrict__ ctx16)
{
  const int node = blockIdx.x * 8 + (threadIdx.x >> 5);
  const int lane = threadIdx.x & 31;
  if (node >= Mrows) return;
  const int b = node / Ss;
  const float ab = alignb[0];

  int nb = -1;
  float scv = -3.0e38f;
  if (lane < KNb) {
    nb  = idx[node * KNb + lane];
    scv = (nb >= 0) ? leakyf(aA[node] + gB[b * Ss + nb] + ab) : NEGV;
  }
  float mx = scv;
#pragma unroll
  for (int o = 16; o; o >>= 1) mx = fmaxf(mx, __shfl_xor(mx, o, 32));
  float e = (lane < KNb) ? __expf(scv - mx) : 0.f;
  float sum = e;
#pragma unroll
  for (int o = 16; o; o >>= 1) sum += __shfl_xor(sum, o, 32);
  const float attw = (lane < KNb && nb >= 0) ? (e / sum) : 0.f;

  float acc[8] = {0.f, 0.f, 0.f, 0.f, 0.f, 0.f, 0.f, 0.f};
  for (int k = 0; k < KNb; ++k) {
    const float wk = __shfl(attw, k, 32);
    const int  nbk = __shfl(nb, k, 32);
    if (nbk >= 0 && wk != 0.f) {
      const float* trow = t + (size_t)(b * Ss + nbk) * Ee;
      __builtin_prefetch(trow, 0, 0);
#pragma unroll
      for (int j = 0; j < 8; ++j) acc[j] += wk * trow[lane + 32 * j];
    }
  }
  _Float16* crow = ctx16 + (size_t)node * Ee;
#pragma unroll
  for (int j = 0; j < 8; ++j) {
    const float v = acc[j];
    crow[lane + 32 * j] = (_Float16)(v > 0.f ? v : (__expf(v) - 1.f));
  }
}

// ---------------------------------------------------------------------------
// Fused GRU gate elementwise: h = (1-z)*n + z*h ; act = relu(h); fp16 mirrors
// ---------------------------------------------------------------------------
__global__ __launch_bounds__(256) void gru_elem_kernel(
    const float* __restrict__ gi, const float* __restrict__ gh,
    float* __restrict__ h, _Float16* __restrict__ h16,
    float* __restrict__ act, _Float16* __restrict__ act16, int Mr)
{
  const int i = blockIdx.x * blockDim.x + threadIdx.x;
  if (i >= Mr * Ee) return;
  const int row = i / Ee, col = i % Ee;
  const float* gir = gi + (size_t)row * 3 * Ee;
  const float* ghr = gh + (size_t)row * 3 * Ee;
  const float ir = gir[col], iz = gir[Ee + col], in = gir[2 * Ee + col];
  const float hr = ghr[col], hz = ghr[Ee + col], hn = ghr[2 * Ee + col];
  const float r = 1.f / (1.f + __expf(-(ir + hr)));
  const float z = 1.f / (1.f + __expf(-(iz + hz)));
  const float n = tanhf(in + r * hn);
  const float hv = h[i];
  const float hnew = (1.f - z) * n + z * hv;
  const float a = hnew > 0.f ? hnew : 0.f;
  h[i] = hnew;   h16[i] = (_Float16)hnew;
  act[i] = a;    act16[i] = (_Float16)a;
}

// ---------------------------------------------------------------------------
// seq_feature[b,:] = sum_s act[b,s,:]*mask[b,s] ; act_mol = relu(seq_feature)
// ---------------------------------------------------------------------------
__global__ __launch_bounds__(256) void seq_pool_kernel(
    const float* __restrict__ act, const float* __restrict__ mask,
    float* __restrict__ seqf, _Float16* __restrict__ seqf16,
    float* __restrict__ amol)
{
  const int b = blockIdx.x, col = threadIdx.x;
  float s = 0.f;
  for (int i = 0; i < Ss; ++i)
    s += act[(size_t)(b * Ss + i) * Ee + col] * mask[b * Ss + i];
  seqf[b * Ee + col]   = s;
  seqf16[b * Ee + col] = (_Float16)s;
  amol[b * Ee + col]   = s > 0.f ? s : 0.f;
}

// ---------------------------------------------------------------------------
// Sequence attention: masked softmax over S=512 + weighted sum of feat_t + elu
// One 256-thread block per batch row; fp16 context output.
// ---------------------------------------------------------------------------
__global__ __launch_bounds__(256) void seq_ctx_kernel(
    const float* __restrict__ aMol, const float* __restrict__ sB,
    const float* __restrict__ salignb, const float* __restrict__ mask,
    const float* __restrict__ feat_t, _Float16* __restrict__ sctx16)
{
  __shared__ float sw[Ss];
  __shared__ float red[8];
  const int b = blockIdx.x;
  const int tid = threadIdx.x, lane = tid & 31, wave = tid >> 5;
  const float am = aMol[b], ab = salignb[0];

  float mx = -3.0e38f;
  for (int s = tid; s < Ss; s += 256) {
    const float mk = mask[b * Ss + s];
    const float sc = (mk != 0.f) ? leakyf(am + sB[b * Ss + s] + ab) : NEGV;
    sw[s] = sc;
    mx = fmaxf(mx, sc);
  }
#pragma unroll
  for (int o = 16; o; o >>= 1) mx = fmaxf(mx, __shfl_xor(mx, o, 32));
  if (lane == 0) red[wave] = mx;
  __syncthreads();
  float bm = red[0];
#pragma unroll
  for (int w = 1; w < 8; ++w) bm = fmaxf(bm, red[w]);
  __syncthreads();

  float sum = 0.f;
  for (int s = tid; s < Ss; s += 256) {
    const float e2 = __expf(sw[s] - bm);   // masked -> exp(NEG-bm) == 0
    sw[s] = e2;
    sum += e2;
  }
#pragma unroll
  for (int o = 16; o; o >>= 1) sum += __shfl_xor(sum, o, 32);
  if (lane == 0) red[wave] = sum;
  __syncthreads();
  float tot = 0.f;
#pragma unroll
  for (int w = 0; w < 8; ++w) tot += red[w];
  __syncthreads();

  float acc = 0.f;
  for (int s = 0; s < Ss; ++s)
    acc += sw[s] * feat_t[(size_t)(b * Ss + s) * Ee + tid];
  acc /= tot;
  sctx16[b * Ee + tid] = (_Float16)(acc > 0.f ? acc : (__expf(acc) - 1.f));
}

// ---------------------------------------------------------------------------
// Host-side orchestration (all on `stream`, graph-capture safe)
// ---------------------------------------------------------------------------
extern "C" void kernel_launch(void* const* d_in, const int* in_sizes, int n_in,
                              void* d_out, int out_size, void* d_ws, size_t ws_size,
                              hipStream_t stream)
{
  (void)in_sizes; (void)n_in; (void)out_size; (void)ws_size;

  const float* amino    = (const float*)d_in[0];
  const int*   idx      = (const int*)  d_in[1];
  const float* mask     = (const float*)d_in[2];
  const float* emb_W    = (const float*)d_in[3];
  const float* emb_b    = (const float*)d_in[4];
  const float* neigh_W  = (const float*)d_in[5];
  const float* neigh_b  = (const float*)d_in[6];
  const float* align_W  = (const float*)d_in[7];   // (R,1,2E)
  const float* align_b  = (const float*)d_in[8];   // (R,1)
  const float* attend_W = (const float*)d_in[9];   // (R,E,E)
  const float* attend_b = (const float*)d_in[10];  // (R,E)
  const float* gru_wih  = (const float*)d_in[11];  // (R,3E,E)
  const float* gru_whh  = (const float*)d_in[12];
  const float* gru_bih  = (const float*)d_in[13];  // (R,3E)
  const float* gru_bhh  = (const float*)d_in[14];
  const float* salign_W = (const float*)d_in[15];  // (1,2E)
  const float* salign_b = (const float*)d_in[16];  // (1,)
  const float* sattend_W= (const float*)d_in[17];  // (E,E)
  const float* sattend_b= (const float*)d_in[18];  // (E,)
  const float* sgru_wih = (const float*)d_in[19];  // (3E,E)
  const float* sgru_whh = (const float*)d_in[20];
  const float* sgru_bih = (const float*)d_in[21];
  const float* sgru_bhh = (const float*)d_in[22];
  float* out = (float*)d_out;
  float* ws  = (float*)d_ws;

  const size_t ME = (size_t)Mrows * Ee;
  // fp32 regions
  float* f_nf   = ws;
  float* f_af   = f_nf   + ME;
  float* f_h    = f_af   + ME;
  float* f_act  = f_h    + ME;
  float* f_t    = f_act  + ME;
  float* f_gi   = f_t    + ME;
  float* f_gh   = f_gi   + ME * 3;
  float* f_aA   = f_gh   + ME * 3;
  float* f_gB   = f_aA   + Mrows;
  float* f_sB   = f_gB   + Mrows;
  float* f_seqf = f_sB   + Mrows;                 // Mpad*Ee (rows 32.. are pad)
  float* f_sgi  = f_seqf + (size_t)Mpad * Ee;     // Mpad*3Ee
  float* f_sgh  = f_sgi  + (size_t)Mpad * 3 * Ee; // Mpad*3Ee
  float* f_amol = f_sgh  + (size_t)Mpad * 3 * Ee; // Bb*Ee
  float* f_aMol = f_amol + Bb * Ee;               // Bb
  // fp16 regions
  _Float16* hbase   = (_Float16*)(f_aMol + Bb);
  _Float16* h_nf    = hbase;
  _Float16* h_af    = h_nf   + ME;
  _Float16* h_h     = h_af   + ME;
  _Float16* h_act   = h_h    + ME;
  _Float16* h_ctx   = h_act  + ME;
  _Float16* h_seqf  = h_ctx  + ME;                     // Mpad*Ee
  _Float16* h_sctx  = h_seqf + (size_t)Mpad * Ee;      // Mpad*Ee
  _Float16* h_dummy = h_sctx + (size_t)Mpad * Ee;      // Bb*Ee (scratch)
  _Float16* h_attW  = h_dummy + Bb * Ee;               // 3*E*E
  _Float16* h_gwih  = h_attW  + 3 * Ee * Ee;           // 3*3E*E
  _Float16* h_gwhh  = h_gwih  + 9 * Ee * Ee;
  _Float16* h_sattW = h_gwhh  + 9 * Ee * Ee;           // E*E
  _Float16* h_swih  = h_sattW + Ee * Ee;               // 3E*E
  _Float16* h_swhh  = h_swih  + 3 * Ee * Ee;

  const dim3 blk(256);
  const dim3 gE(Mrows / TM, Ee / TN);        // (128, 2)
  const dim3 g3E(Mrows / TM, (3 * Ee) / TN); // (128, 6)
  const dim3 gSeq3E(1, (3 * Ee) / TN);       // (1, 6)

  // one-time fp16 weight mirrors (cheap, deterministic)
  const int nAtt = 3 * Ee * Ee / 4, nGru = 9 * Ee * Ee / 4;
  const int nSAtt = Ee * Ee / 4, nSGru = 3 * Ee * Ee / 4;
  cvt16_kernel<<<(nAtt + 255) / 256, blk, 0, stream>>>(attend_W, h_attW, nAtt);
  cvt16_kernel<<<(nGru + 255) / 256, blk, 0, stream>>>(gru_wih, h_gwih, nGru);
  cvt16_kernel<<<(nGru + 255) / 256, blk, 0, stream>>>(gru_whh, h_gwhh, nGru);
  cvt16_kernel<<<(nSAtt + 255) / 256, blk, 0, stream>>>(sattend_W, h_sattW, nSAtt);
  cvt16_kernel<<<(nSGru + 255) / 256, blk, 0, stream>>>(sgru_wih, h_swih, nSGru);
  cvt16_kernel<<<(nSGru + 255) / 256, blk, 0, stream>>>(sgru_whh, h_swhh, nSGru);

  // embeddings: gather commutes with row-wise linear+leaky, so compute per-row
  gemm26_f16_wmma<<<gE, blk, 0, stream>>>(amino, neigh_W, neigh_b, f_nf, h_nf);
  gemm26_f16_wmma<<<gE, blk, 0, stream>>>(amino, emb_W,   emb_b,   f_af, h_af);
  hipMemcpyAsync(f_h, f_af, ME * sizeof(float), hipMemcpyDeviceToDevice, stream);
  hipMemcpyAsync(h_h, h_af, ME * sizeof(_Float16), hipMemcpyDeviceToDevice, stream);

  for (int d = 0; d < 3; ++d) {
    const float*    src   = (d == 0) ? f_nf : f_act;   // rows that get gathered
    const float*    cur   = (d == 0) ? f_af : f_act;   // query rows
    const _Float16* src16 = (d == 0) ? h_nf : h_act;
    gemm256h_f16_wmma<<<gE, blk, 0, stream>>>(src16, h_attW + (size_t)d * Ee * Ee,
                                              attend_b + d * Ee, f_t, Ee);
    rowdot_kernel<<<(Mrows + 7) / 8, blk, 0, stream>>>(cur, align_W + (size_t)d * 2 * Ee,      f_aA, Mrows);
    rowdot_kernel<<<(Mrows + 7) / 8, blk, 0, stream>>>(src, align_W + (size_t)d * 2 * Ee + Ee, f_gB, Mrows);
    attend_ctx_kernel<<<Mrows / 8, blk, 0, stream>>>(idx, f_aA, f_gB, align_b + d, f_t, h_ctx);
    gemm256h_f16_wmma<<<g3E, blk, 0, stream>>>(h_ctx, h_gwih + (size_t)d * 3 * Ee * Ee,
                                               gru_bih + d * 3 * Ee, f_gi, 3 * Ee);
    gemm256h_f16_wmma<<<g3E, blk, 0, stream>>>(h_h, h_gwhh + (size_t)d * 3 * Ee * Ee,
                                               gru_bhh + d * 3 * Ee, f_gh, 3 * Ee);
    gru_elem_kernel<<<(Mrows * Ee + 255) / 256, blk, 0, stream>>>(f_gi, f_gh,
                                               f_h, h_h, f_act, h_act, Mrows);
  }

  // sequence head. act_mol is loop-invariant in the reference T-loop, so the
  // scores / context / gi are computed once; only the GRU gh recurrence iterates.
  seq_pool_kernel<<<Bb, blk, 0, stream>>>(f_act, mask, f_seqf, h_seqf, f_amol);
  gemm256h_f16_wmma<<<gE, blk, 0, stream>>>(h_act, h_sattW, sattend_b, f_t, Ee);
  rowdot_kernel<<<(Mrows + 7) / 8, blk, 0, stream>>>(f_act, salign_W + Ee, f_sB, Mrows);
  rowdot_kernel<<<(Bb + 7) / 8, blk, 0, stream>>>(f_amol, salign_W, f_aMol, Bb);
  seq_ctx_kernel<<<Bb, blk, 0, stream>>>(f_aMol, f_sB, salign_b, mask, f_t, h_sctx);
  gemm256h_f16_wmma<<<gSeq3E, blk, 0, stream>>>(h_sctx, h_swih, sgru_bih, f_sgi, 3 * Ee);
  for (int t = 0; t < 2; ++t) {
    gemm256h_f16_wmma<<<gSeq3E, blk, 0, stream>>>(h_seqf, h_swhh, sgru_bhh, f_sgh, 3 * Ee);
    gru_elem_kernel<<<(Bb * Ee + 255) / 256, blk, 0, stream>>>(f_sgi, f_sgh,
                                               f_seqf, h_seqf, out, h_dummy, Bb);
  }
}
